// NCHM_26422638805482
// MI455X (gfx1250) — matrix operand order repaired and checked by reading, not compiled, not run on hardware
//
#include <hip/hip_runtime.h>
#include <hip/hip_bf16.h>
#include <math.h>

// ---------------- problem constants (from reference) ----------------
#define HEADS   8
#define NHID    128
#define NLAST   64
#define MNODES  400
#define EN      256
#define NEDGEI  4096       // incidence count (hyperedge_index columns)
#define NNODES  512        // N
#define PPAIRS  2048       // P
#define DDOCS   64
#define LSENT   256
#define HDIM    768
#define NEG_SLOPE 0.2f

// ---------------- WMMA types ----------------
typedef __attribute__((ext_vector_type(16))) __bf16 v16bf;
typedef __attribute__((ext_vector_type(8)))  float  v8f;

// ---------------- workspace layout (float offsets) ----------------
__host__ __device__ constexpr size_t padf(size_t n) { return (n + 63) & ~size_t(63); }
constexpr size_t OFF_MERGED = 0;                                    // 400*768
constexpr size_t OFF_CNT    = OFF_MERGED + padf(400 * 768);         // 400
constexpr size_t OFF_EATTR1 = OFF_CNT    + padf(400);               // 256*768
constexpr size_t OFF_DEGD   = OFF_EATTR1 + padf(256 * 768);         // 400
constexpr size_t OFF_DEGB   = OFF_DEGD   + padf(400);               // 256
constexpr size_t OFF_XW1    = OFF_DEGB   + padf(256);               // 400*1024
constexpr size_t OFF_EW1    = OFF_XW1    + padf(400 * 1024);        // 256*1024
constexpr size_t OFF_ALPHA  = OFF_EW1    + padf(256 * 1024);        // 4096*8
constexpr size_t OFF_AMAX   = OFF_ALPHA  + padf(4096 * 8);          // 400*8
constexpr size_t OFF_DEN    = OFF_AMAX   + padf(400 * 8);           // 400*8
constexpr size_t OFF_EOUT   = OFF_DEN    + padf(400 * 8);           // 256*1024 (reused layer2)
constexpr size_t OFF_NACC   = OFF_EOUT   + padf(256 * 1024);        // 400*1024 -> h1
constexpr size_t OFF_EATTR2 = OFF_NACC   + padf(400 * 1024);        // 256*1024
constexpr size_t OFF_XW2    = OFF_EATTR2 + padf(256 * 1024);        // 400*512
constexpr size_t OFF_EW2    = OFF_XW2    + padf(400 * 512);         // 256*512
constexpr size_t OFF_H2     = OFF_EW2    + padf(256 * 512);         // 400*512
constexpr size_t OFF_FEAT   = OFF_H2     + padf(400 * 512);         // 2048*2560
constexpr size_t OFF_HID    = OFF_FEAT   + padf(2048 * 2560);       // 2048*1024

// ---------------- small helpers ----------------
__global__ void fill_f32(float* p, int n, float v) {
    int t = blockIdx.x * blockDim.x + threadIdx.x;
    if (t < n) p[t] = v;
}

__device__ inline void atomicMaxF(float* addr, float val) {
    unsigned int* ua = (unsigned int*)addr;
    unsigned int old = *ua;
    while (true) {
        float f = __uint_as_float(old);
        if (f >= val) break;
        unsigned int assumed = old;
        old = atomicCAS(ua, assumed, __float_as_uint(val));
        if (old == assumed) break;
    }
}

__device__ inline unsigned int pack_bf16(float lo, float hi) {
    union { unsigned int u; __bf16 h[2]; } pk;
    pk.h[0] = (__bf16)lo;
    pk.h[1] = (__bf16)hi;
    return pk.u;
}

// =====================================================================
// LDS-tiled WMMA GEMM: C[M,N] = A[M,K] @ B[K,N] (+bias, +relu)
//  - block: 128 threads = 4 waves (wave32); block tile 64(M) x 64(N), K-step 32
//  - A/B tiles staged through LDS, converted fp32 -> bf16 once per tile
//    (loads issued in a batch first so 8+ global loads stay in flight,
//     then v_cvt_pk_bf16_f32 + ds_store_b32)
//  - LDS layouts are "pair-packed" so every fragment dword is ONE ds_load_b32:
//      A: row-major bf16, K-adjacent pairs are adjacent (natural layout)
//      B: BsP[p][n] packs {B[2p][n], B[2p+1][n]} in one dword
//  - each wave: 1 A-fragment (rows w*16..w*16+15) reused across 4 B-fragments
//    -> 4 v_wmma_f32_16x16x32_bf16 per K-step, fp32 accumulate
//  - __builtin_prefetch on next K-tile -> global_prefetch_b8
// Fragment layouts per ISA 7.12.2:
//  A 16x32 bf16: lane m=lane&15, kh=lane>>4; vgpr v: K=(v<4?0:16)+kh*8+(v&3)*2
//  B 32x16 bf16: lane n=lane&15, kh=lane>>4; vgpr v: K=(v<4?0:16)+kh*8+(v&3)*2
//  C/D f32 16x16: lane n=lane&15; vgpr v: row M=v+8*(lane>>4)
// =====================================================================
#define GT_M 64
#define GT_N 64
#define GT_K 32
#define A_STRIDE 36    // halves per LDS A row (32 + pad) -> dword stride 18
#define BP_STRIDE 68   // dwords per LDS B pair-row (64 + pad)

__global__ __launch_bounds__(128)
void wmma_gemm_bf16(const float* __restrict__ A, const float* __restrict__ Bm,
                    const float* __restrict__ bias, float* __restrict__ C,
                    int Mrows, int K, int Ncols, int relu) {
    __shared__ __bf16       As[GT_M * A_STRIDE];     // 4608 B
    __shared__ unsigned int BsP[16 * BP_STRIDE];     // 4352 B (16 k-pairs x 64 cols)

    const int tid  = threadIdx.x;
    const int w    = tid >> 5;          // wave 0..3 -> M sub-tile
    const int lane = tid & 31;
    const int ln   = lane & 15;
    const int kh   = lane >> 4;
    const int m0   = blockIdx.y * GT_M;
    const int n0   = blockIdx.x * GT_N;

    v8f acc0 = {}, acc1 = {}, acc2 = {}, acc3 = {};

    for (int k0 = 0; k0 < K; k0 += GT_K) {
        // ---- stage A tile (64 x 32): batch-load 8 float2, then cvt+store ----
        float2 areg[8];
#pragma unroll
        for (int i = 0; i < 8; ++i) {
            const int idx = tid + i * 128;
            const int row = idx >> 4;          // 0..63
            const int kp  = idx & 15;          // pair index 0..15
            const int gr  = (m0 + row < Mrows) ? (m0 + row) : (Mrows - 1);
            areg[i] = *reinterpret_cast<const float2*>(A + (long)gr * K + k0 + 2 * kp);
        }
        // ---- stage B tile (32 x 64) pair-packed: two strided loads per unit ----
        float blo[8], bhi[8];
#pragma unroll
        for (int i = 0; i < 8; ++i) {
            const int idx = tid + i * 128;
            const int p   = idx >> 6;          // k-pair 0..15
            const int n   = idx & 63;          // column 0..63
            const long base = (long)(k0 + 2 * p) * Ncols + n0 + n;
            blo[i] = Bm[base];
            bhi[i] = Bm[base + Ncols];
        }
#pragma unroll
        for (int i = 0; i < 8; ++i) {
            const int idx = tid + i * 128;
            const int row = idx >> 4, kp = idx & 15;
            *reinterpret_cast<unsigned int*>(&As[row * A_STRIDE + 2 * kp]) =
                pack_bf16(areg[i].x, areg[i].y);
        }
#pragma unroll
        for (int i = 0; i < 8; ++i) {
            const int idx = tid + i * 128;
            const int p = idx >> 6, n = idx & 63;
            BsP[p * BP_STRIDE + n] = pack_bf16(blo[i], bhi[i]);
        }
        __syncthreads();

        // prefetch next K-tile into cache while we compute (global_prefetch_b8)
        if (k0 + GT_K < K) {
            const int pr = (m0 + (tid >> 1) < Mrows) ? (m0 + (tid >> 1)) : (Mrows - 1);
            __builtin_prefetch(&A[(long)pr * K + k0 + GT_K + (tid & 1) * 16], 0, 1);
            __builtin_prefetch(&Bm[(long)(k0 + GT_K + (tid >> 2)) * Ncols + n0 + (tid & 3) * 16], 0, 1);
        }

        // ---- build fragments from LDS (one ds_load_b32 per fragment dword) ----
        union { v16bf v; unsigned int u[8]; } af, bf0, bf1, bf2, bf3;
#pragma unroll
        for (int v = 0; v < 8; ++v) {
            const int kb = ((v < 4) ? 0 : 16) + kh * 8 + (v & 3) * 2;   // even
            af.u[v] = *reinterpret_cast<const unsigned int*>(
                          &As[(w * 16 + ln) * A_STRIDE + kb]);
        }
#pragma unroll
        for (int v = 0; v < 8; ++v) {
            const int pr = (((v < 4) ? 0 : 8) + kh * 4 + (v & 3)) * BP_STRIDE + ln;   // k-pair row, col ln
            bf0.u[v] = BsP[pr];
            bf1.u[v] = BsP[pr + 16];
            bf2.u[v] = BsP[pr + 32];
            bf3.u[v] = BsP[pr + 48];
        }
        acc0 = __builtin_amdgcn_wmma_f32_16x16x32_bf16(false, af.v, false, bf0.v, (short)0, acc0, false, false);
        acc1 = __builtin_amdgcn_wmma_f32_16x16x32_bf16(false, af.v, false, bf1.v, (short)0, acc1, false, false);
        acc2 = __builtin_amdgcn_wmma_f32_16x16x32_bf16(false, af.v, false, bf2.v, (short)0, acc2, false, false);
        acc3 = __builtin_amdgcn_wmma_f32_16x16x32_bf16(false, af.v, false, bf3.v, (short)0, acc3, false, false);
        __syncthreads();
    }

    // ---- epilogue: bias + optional relu, guarded store ----
#define GEMM_STORE(ACC, G)                                                     \
    {                                                                          \
        const int gn = n0 + (G) * 16 + ln;                                     \
        const float bv = bias ? bias[gn] : 0.0f;                               \
        _Pragma("unroll")                                                      \
        for (int v = 0; v < 8; ++v) {                                          \
            const int row = m0 + w * 16 + v + kh * 8;                          \
            if (row < Mrows) {                                                 \
                float val = (ACC)[v] + bv;                                     \
                if (relu) val = fmaxf(val, 0.0f);                              \
                C[(long)row * Ncols + gn] = val;                               \
            }                                                                  \
        }                                                                      \
    }
    GEMM_STORE(acc0, 0)
    GEMM_STORE(acc1, 1)
    GEMM_STORE(acc2, 2)
    GEMM_STORE(acc3, 3)
#undef GEMM_STORE
}

// ---------------- graph plumbing kernels (fp32 VALU + atomics) ----------------

// merged_sum[co_id[n]] += node_s_emb[n, node_e[n], :]; cnt[co]++
__global__ void node_merge_kernel(const float* __restrict__ node_s_emb,
                                  const int* __restrict__ node_e,
                                  const int* __restrict__ co_id,
                                  float* __restrict__ msum, float* __restrict__ cnt) {
    const int n  = blockIdx.x;                 // 0..511
    const int co = co_id[n];
    const int l  = node_e[n];
    const float* src = node_s_emb + ((long)n * LSENT + l) * HDIM;
    float* dst = msum + (long)co * HDIM;
    for (int c = threadIdx.x; c < HDIM; c += blockDim.x)
        atomicAdd(&dst[c], src[c]);
    if (threadIdx.x == 0) atomicAdd(&cnt[co], 1.0f);
}

// x[r][c] /= max(cnt[r],1)
__global__ void divide_rows(float* __restrict__ x, const float* __restrict__ cnt,
                            int rows, int C) {
    long t = (long)blockIdx.x * blockDim.x + threadIdx.x;
    if (t >= (long)rows * C) return;
    int r = (int)(t / C);
    x[t] = x[t] / fmaxf(cnt[r], 1.0f);
}

// node degrees D over n_idx, edge degrees B over e_idx
__global__ void degrees_kernel(const int* __restrict__ n_idx, const int* __restrict__ e_idx,
                               float* __restrict__ D, float* __restrict__ B) {
    int j = blockIdx.x * blockDim.x + threadIdx.x;
    if (j >= NEDGEI) return;
    atomicAdd(&D[n_idx[j]], 1.0f);
    atomicAdd(&B[e_idx[j]], 1.0f);
}

// esum[e_idx[j]] += x[n_idx[j]]  (edge_mean numerator)
__global__ void edge_scatter_sum(const float* __restrict__ x,
                                 const int* __restrict__ n_idx, const int* __restrict__ e_idx,
                                 float* __restrict__ esum, int C) {
    const int j = blockIdx.x;
    const float* src = x + (long)n_idx[j] * C;
    float* dst = esum + (long)e_idx[j] * C;
    for (int c = threadIdx.x; c < C; c += blockDim.x)
        atomicAdd(&dst[c], src[c]);
}

// per-(incidence, head): alpha = leaky_relu(att . [xw[n], ew[e]]); atomicMax amax[n]
__global__ void attn_alpha_kernel(const float* __restrict__ xw, const float* __restrict__ ew,
                                  const int* __restrict__ n_idx, const int* __restrict__ e_idx,
                                  const float* __restrict__ att,
                                  float* __restrict__ alpha, float* __restrict__ amax, int C) {
    int t = blockIdx.x * blockDim.x + threadIdx.x;
    if (t >= NEDGEI * HEADS) return;
    const int j = t >> 3, h = t & 7;
    const int n = n_idx[j], e = e_idx[j];
    const float* xr = xw + ((long)n * HEADS + h) * C;
    const float* er = ew + ((long)e * HEADS + h) * C;
    const float* at = att + (long)h * 2 * C;
    float s = 0.0f;
    for (int c = 0; c < C; ++c) s += at[c] * xr[c];
    for (int c = 0; c < C; ++c) s += at[C + c] * er[c];
    s = (s > 0.0f) ? s : NEG_SLOPE * s;
    alpha[t] = s;
    atomicMaxF(&amax[n * HEADS + h], s);
}

__global__ void sanitize_amax(float* __restrict__ amax, int n) {
    int t = blockIdx.x * blockDim.x + threadIdx.x;
    if (t >= n) return;
    float v = amax[t];
    if (!(fabsf(v) < INFINITY)) amax[t] = 0.0f;
}

__global__ void softmax_ex_kernel(float* __restrict__ alpha, const int* __restrict__ n_idx,
                                  const float* __restrict__ amax, float* __restrict__ den) {
    int t = blockIdx.x * blockDim.x + threadIdx.x;
    if (t >= NEDGEI * HEADS) return;
    const int j = t >> 3, h = t & 7;
    const int n = n_idx[j];
    float ex = expf(alpha[t] - amax[n * HEADS + h]);
    alpha[t] = ex;
    atomicAdd(&den[n * HEADS + h], ex);
}

__global__ void softmax_norm_kernel(float* __restrict__ alpha, const int* __restrict__ n_idx,
                                    const float* __restrict__ den) {
    int t = blockIdx.x * blockDim.x + threadIdx.x;
    if (t >= NEDGEI * HEADS) return;
    const int j = t >> 3, h = t & 7;
    const int n = n_idx[j];
    alpha[t] = alpha[t] / fmaxf(den[n * HEADS + h], 1e-16f);
}

// eout[e] += Binv[e]*alpha[j,h]*xw[n]  (per incidence j)
__global__ void msg1_kernel(const float* __restrict__ xw, const float* __restrict__ alpha,
                            const int* __restrict__ n_idx, const int* __restrict__ e_idx,
                            const float* __restrict__ B, float* __restrict__ eout, int C) {
    const int j = blockIdx.x;
    const int n = n_idx[j], e = e_idx[j];
    const float bc = B[e];
    const float binv = (bc > 0.0f) ? (1.0f / bc) : 0.0f;
    const int HC = HEADS * C;
    const float* src = xw + (long)n * HC;
    float* dst = eout + (long)e * HC;
    for (int hc = threadIdx.x; hc < HC; hc += blockDim.x) {
        const int h = hc / C;
        atomicAdd(&dst[hc], binv * alpha[j * HEADS + h] * src[hc]);
    }
}

// nacc[n] += Dinv[n]*alpha[j,h]*eout[e]
__global__ void msg2_kernel(const float* __restrict__ eout, const float* __restrict__ alpha,
                            const int* __restrict__ n_idx, const int* __restrict__ e_idx,
                            const float* __restrict__ D, float* __restrict__ nacc, int C) {
    const int j = blockIdx.x;
    const int n = n_idx[j], e = e_idx[j];
    const float dc = D[n];
    const float dinv = (dc > 0.0f) ? (1.0f / dc) : 0.0f;
    const int HC = HEADS * C;
    const float* src = eout + (long)e * HC;
    float* dst = nacc + (long)n * HC;
    for (int hc = threadIdx.x; hc < HC; hc += blockDim.x) {
        const int h = hc / C;
        atomicAdd(&dst[hc], dinv * alpha[j * HEADS + h] * src[hc]);
    }
}

__global__ void bias_act_kernel(float* __restrict__ x, const float* __restrict__ bias,
                                long total, int C, int relu) {
    long t = (long)blockIdx.x * blockDim.x + threadIdx.x;
    if (t >= total) return;
    float v = x[t] + bias[t % C];
    if (relu) v = fmaxf(v, 0.0f);
    x[t] = v;
}

// feat[p] = [a+b | a-b | e1 | e2]
__global__ void build_feat_kernel(const float* __restrict__ h2, const int* __restrict__ co_id,
                                  const int* __restrict__ event_id,
                                  const float* __restrict__ sent_emb,
                                  const int* __restrict__ pair_doc,
                                  const int* __restrict__ event_e,
                                  float* __restrict__ feat) {
    const int p  = blockIdx.x;
    const int na = co_id[event_id[2 * p]];
    const int nb = co_id[event_id[2 * p + 1]];
    const float* ha = h2 + (long)na * (HEADS * NLAST);
    const float* hb = h2 + (long)nb * (HEADS * NLAST);
    const int d = pair_doc[p];
    const float* e1 = sent_emb + ((long)d * LSENT + event_e[2 * p]) * HDIM;
    const float* e2 = sent_emb + ((long)d * LSENT + event_e[2 * p + 1]) * HDIM;
    float* fr = feat + (long)p * 2560;
    for (int c = threadIdx.x; c < HEADS * NLAST; c += blockDim.x) {
        const float a = ha[c], b = hb[c];
        fr[c]       = a + b;
        fr[512 + c] = a - b;
    }
    for (int c = threadIdx.x; c < HDIM; c += blockDim.x) {
        fr[1024 + c] = e1[c];
        fr[1792 + c] = e2[c];
    }
}

// out[p, o] = hid[p] . Wm2[:, o] + bm2[o]   (N=2 -> plain dot, not worth WMMA)
__global__ void out_small_kernel(const float* __restrict__ hid, const float* __restrict__ Wm2,
                                 const float* __restrict__ bm2, float* __restrict__ out) {
    int t = blockIdx.x * blockDim.x + threadIdx.x;
    if (t >= PPAIRS * 2) return;
    const int p = t >> 1, o = t & 1;
    const float* hr = hid + (long)p * 1024;
    float s = bm2[o];
    for (int k = 0; k < 1024; ++k) s += hr[k] * Wm2[k * 2 + o];
    out[t] = s;
}

// ---------------- host launcher ----------------
static inline dim3 g1(long n, int b) { return dim3((unsigned)((n + b - 1) / b)); }
static inline unsigned cdiv(int a, int b) { return (unsigned)((a + b - 1) / b); }

extern "C" void kernel_launch(void* const* d_in, const int* in_sizes, int n_in,
                              void* d_out, int out_size, void* d_ws, size_t ws_size,
                              hipStream_t stream) {
    const float* sent_emb   = (const float*)d_in[0];
    const float* node_s_emb = (const float*)d_in[1];
    const float* W1   = (const float*)d_in[2];
    const float* att1 = (const float*)d_in[3];
    const float* b1   = (const float*)d_in[4];
    const float* W2   = (const float*)d_in[5];
    const float* att2 = (const float*)d_in[6];
    const float* b2   = (const float*)d_in[7];
    const float* Wm1  = (const float*)d_in[8];
    const float* bm1  = (const float*)d_in[9];
    const float* Wm2  = (const float*)d_in[10];
    const float* bm2  = (const float*)d_in[11];
    const int* pair_doc  = (const int*)d_in[12];
    const int* event_e   = (const int*)d_in[13];
    const int* node_e    = (const int*)d_in[14];
    const int* co_id     = (const int*)d_in[15];
    const int* hyperedge = (const int*)d_in[16];   // [2,4096]
    const int* event_id  = (const int*)d_in[17];
    const int* n_idx = hyperedge;
    const int* e_idx = hyperedge + NEDGEI;

    float* ws = (float*)d_ws;
    float* merged = ws + OFF_MERGED;
    float* cnt    = ws + OFF_CNT;
    float* eattr1 = ws + OFF_EATTR1;
    float* D      = ws + OFF_DEGD;
    float* B      = ws + OFF_DEGB;
    float* xw1    = ws + OFF_XW1;
    float* ew1    = ws + OFF_EW1;
    float* alpha  = ws + OFF_ALPHA;
    float* amax   = ws + OFF_AMAX;
    float* den    = ws + OFF_DEN;
    float* eout   = ws + OFF_EOUT;
    float* nacc   = ws + OFF_NACC;      // becomes h1 in place
    float* eattr2 = ws + OFF_EATTR2;
    float* xw2    = ws + OFF_XW2;
    float* ew2    = ws + OFF_EW2;
    float* h2     = ws + OFF_H2;
    float* feat   = ws + OFF_FEAT;
    float* hid    = ws + OFF_HID;
    float* out    = (float*)d_out;

    // ---- zero / init accumulators ----
    fill_f32<<<g1(400 * 768, 256), 256, 0, stream>>>(merged, 400 * 768, 0.0f);
    fill_f32<<<g1(400, 256), 256, 0, stream>>>(cnt, 400, 0.0f);
    fill_f32<<<g1(256 * 768, 256), 256, 0, stream>>>(eattr1, 256 * 768, 0.0f);
    fill_f32<<<g1(400, 256), 256, 0, stream>>>(D, 400, 0.0f);
    fill_f32<<<g1(256, 256), 256, 0, stream>>>(B, 256, 0.0f);
    fill_f32<<<g1(400 * 8, 256), 256, 0, stream>>>(amax, 400 * 8, -INFINITY);
    fill_f32<<<g1(400 * 8, 256), 256, 0, stream>>>(den, 400 * 8, 0.0f);
    fill_f32<<<g1(256 * 1024, 256), 256, 0, stream>>>(eout, 256 * 1024, 0.0f);
    fill_f32<<<g1(400 * 1024, 256), 256, 0, stream>>>(nacc, 400 * 1024, 0.0f);
    fill_f32<<<g1(256 * 1024, 256), 256, 0, stream>>>(eattr2, 256 * 1024, 0.0f);
    fill_f32<<<g1(400 * 512, 256), 256, 0, stream>>>(h2, 400 * 512, 0.0f);

    // ---- merged node features (segment mean over co_id) ----
    node_merge_kernel<<<NNODES, 256, 0, stream>>>(node_s_emb, node_e, co_id, merged, cnt);
    divide_rows<<<g1((long)400 * 768, 256), 256, 0, stream>>>(merged, cnt, 400, 768);
    degrees_kernel<<<g1(NEDGEI, 256), 256, 0, stream>>>(n_idx, e_idx, D, B);

    // ---- layer 1: eattr1 = edge_mean(merged) ----
    edge_scatter_sum<<<NEDGEI, 256, 0, stream>>>(merged, n_idx, e_idx, eattr1, 768);
    divide_rows<<<g1((long)256 * 768, 256), 256, 0, stream>>>(eattr1, B, 256, 768);

    // GEMMs (LDS-tiled WMMA bf16): xw1 = merged@W1, ew1 = eattr1@W1
    wmma_gemm_bf16<<<dim3(1024 / GT_N, cdiv(400, GT_M)), 128, 0, stream>>>(merged, W1, nullptr, xw1, 400, 768, 1024, 0);
    wmma_gemm_bf16<<<dim3(1024 / GT_N, cdiv(256, GT_M)), 128, 0, stream>>>(eattr1, W1, nullptr, ew1, 256, 768, 1024, 0);

    // attention + segment softmax over n_idx
    attn_alpha_kernel<<<g1(NEDGEI * HEADS, 256), 256, 0, stream>>>(xw1, ew1, n_idx, e_idx, att1, alpha, amax, NHID);
    sanitize_amax<<<g1(400 * 8, 256), 256, 0, stream>>>(amax, 400 * 8);
    softmax_ex_kernel<<<g1(NEDGEI * HEADS, 256), 256, 0, stream>>>(alpha, n_idx, amax, den);
    softmax_norm_kernel<<<g1(NEDGEI * HEADS, 256), 256, 0, stream>>>(alpha, n_idx, den);

    // message passing
    msg1_kernel<<<NEDGEI, 256, 0, stream>>>(xw1, alpha, n_idx, e_idx, B, eout, NHID);
    msg2_kernel<<<NEDGEI, 256, 0, stream>>>(eout, alpha, n_idx, e_idx, D, nacc, NHID);
    bias_act_kernel<<<g1((long)400 * 1024, 256), 256, 0, stream>>>(nacc, b1, (long)400 * 1024, 1024, 1);
    float* h1 = nacc;

    // ---- layer 2 ----
    edge_scatter_sum<<<NEDGEI, 256, 0, stream>>>(h1, n_idx, e_idx, eattr2, 1024);
    divide_rows<<<g1((long)256 * 1024, 256), 256, 0, stream>>>(eattr2, B, 256, 1024);

    wmma_gemm_bf16<<<dim3(512 / GT_N, cdiv(400, GT_M)), 128, 0, stream>>>(h1, W2, nullptr, xw2, 400, 1024, 512, 0);
    wmma_gemm_bf16<<<dim3(512 / GT_N, cdiv(256, GT_M)), 128, 0, stream>>>(eattr2, W2, nullptr, ew2, 256, 1024, 512, 0);

    // reset softmax / eout accumulators for layer 2
    fill_f32<<<g1(400 * 8, 256), 256, 0, stream>>>(amax, 400 * 8, -INFINITY);
    fill_f32<<<g1(400 * 8, 256), 256, 0, stream>>>(den, 400 * 8, 0.0f);
    fill_f32<<<g1(256 * 512, 256), 256, 0, stream>>>(eout, 256 * 512, 0.0f);

    attn_alpha_kernel<<<g1(NEDGEI * HEADS, 256), 256, 0, stream>>>(xw2, ew2, n_idx, e_idx, att2, alpha, amax, NLAST);
    sanitize_amax<<<g1(400 * 8, 256), 256, 0, stream>>>(amax, 400 * 8);
    softmax_ex_kernel<<<g1(NEDGEI * HEADS, 256), 256, 0, stream>>>(alpha, n_idx, amax, den);
    softmax_norm_kernel<<<g1(NEDGEI * HEADS, 256), 256, 0, stream>>>(alpha, n_idx, den);

    msg1_kernel<<<NEDGEI, 256, 0, stream>>>(xw2, alpha, n_idx, e_idx, B, eout, NLAST);
    msg2_kernel<<<NEDGEI, 256, 0, stream>>>(eout, alpha, n_idx, e_idx, D, h2, NLAST);
    bias_act_kernel<<<g1((long)400 * 512, 256), 256, 0, stream>>>(h2, b2, (long)400 * 512, 512, 0);

    // ---- MLP head ----
    build_feat_kernel<<<PPAIRS, 256, 0, stream>>>(h2, co_id, event_id, sent_emb, pair_doc, event_e, feat);
    // dominant GEMM: feat(2048x2560) @ Wm1(2560x1024) + bm1, relu  (WMMA)
    wmma_gemm_bf16<<<dim3(1024 / GT_N, cdiv(2048, GT_M)), 128, 0, stream>>>(feat, Wm1, bm1, hid, 2048, 2560, 1024, 1);
    out_small_kernel<<<g1(PPAIRS * 2, 256), 256, 0, stream>>>(hid, Wm2, bm2, out);
}
